// AgentSelfAttention_55207509623099
// MI455X (gfx1250) — matrix operand order repaired and checked by reading, not compiled, not run on hardware
//
#include <hip/hip_runtime.h>
#include <hip/hip_bf16.h>

#define DEV __device__ __forceinline__

typedef __bf16 bf16_t;
typedef __attribute__((ext_vector_type(16))) __bf16 v16bf;
typedef __attribute__((ext_vector_type(8)))  __bf16 v8bf;
typedef __attribute__((ext_vector_type(8)))  float  v8f;

static constexpr int Bsz = 4, Nsz = 4096, DIMsz = 1024, Hsz = 16, Dsz = 64, Msz = 128;
static constexpr int DINNER = Hsz * Dsz;   // 1024
static constexpr int QKV3 = 3 * DINNER;    // 3072
#define SCALE_F 0.125f                     // 64^-0.5

// padded LDS strides (bf16 elems, multiple of 8 => 16B-aligned rows, non-pow2 => fewer bank conflicts)
static constexpr int S40 = 40, S72 = 72, S136 = 136;

// ---- bf16 helpers (round-to-nearest-even) ----
DEV bf16_t f2bf(float f) {
  unsigned u = __builtin_bit_cast(unsigned, f);
  unsigned r = u + 0x7FFFu + ((u >> 16) & 1u);
  unsigned short h = (unsigned short)(r >> 16);
  return __builtin_bit_cast(bf16_t, h);
}
DEV float bf2f(bf16_t b) {
  unsigned short h = __builtin_bit_cast(unsigned short, b);
  unsigned u = ((unsigned)h) << 16;
  return __builtin_bit_cast(float, u);
}
DEV void cp16(bf16_t* dst, const bf16_t* src) {   // 16-byte vector copy
  *(v8bf*)dst = *(const v8bf*)src;
}

// ---- WMMA fragment loaders: each = two ds_load_b128 ----
// A: 16 rows x 32 K, LDS row-major [row][k], stride ld.
DEV v16bf frag_a(const bf16_t* s, int ld) {
  const int lane = threadIdx.x & 31;
  const bf16_t* p = s + (lane & 15) * ld + ((lane >> 4) << 3);
  v8bf lo = *(const v8bf*)p;
  v8bf hi = *(const v8bf*)(p + 16);
  return __builtin_shufflevector(lo, hi, 0,1,2,3,4,5,6,7,8,9,10,11,12,13,14,15);
}
// B: 32 K x 16 cols, staged N-major in LDS: sT[col][k], stride ld.
DEV v16bf frag_bt(const bf16_t* s, int ld) {
  const int lane = threadIdx.x & 31;
  const bf16_t* p = s + (lane & 15) * ld + ((lane >> 4) << 3);
  v8bf lo = *(const v8bf*)p;
  v8bf hi = *(const v8bf*)(p + 16);
  return __builtin_shufflevector(lo, hi, 0,1,2,3,4,5,6,7,8,9,10,11,12,13,14,15);
}
DEV v8f wmma_bf(v16bf a, v16bf b, v8f c) {
  return __builtin_amdgcn_wmma_f32_16x16x32_bf16(false, a, false, b, (short)0, c, false, false);
}

// ============ Kernel 1: qkv = x @ W_qkv, split into q/k/v bf16 [b,h,n,d] ============
__global__ void __launch_bounds__(128)
k_qkv(const float* __restrict__ x, const float* __restrict__ wqkv,
      bf16_t* __restrict__ qb, bf16_t* __restrict__ kb, bf16_t* __restrict__ vb) {
  __shared__ alignas(16) bf16_t sx[64][S40];    // A tile rows x K(32)
  __shared__ alignas(16) bf16_t swT[64][S40];   // B tile N-major: [col][k]
  const int row0 = blockIdx.x * 64;   // over B*N
  const int col0 = blockIdx.y * 64;   // over 3072
  const int tid = threadIdx.x, wave = tid >> 5, lane = tid & 31;
  v8f acc[4] = {};
  for (int k0 = 0; k0 < DIMsz; k0 += 32) {
    for (int i = tid; i < 64 * 32; i += 128) {            // convert f32 -> bf16
      int r = i >> 5, c = i & 31;
      sx[r][c] = f2bf(x[(size_t)(row0 + r) * DIMsz + k0 + c]);
    }
    for (int i = tid; i < 32 * 64; i += 128) {            // coalesced read, transpose into LDS
      int r = i >> 6, c = i & 63;
      swT[c][r] = f2bf(wqkv[(size_t)(k0 + r) * QKV3 + col0 + c]);
    }
    __syncthreads();
    v16bf af = frag_a(&sx[wave * 16][0], S40);
#pragma unroll
    for (int t = 0; t < 4; ++t) acc[t] = wmma_bf(af, frag_bt(&swT[t * 16][0], S40), acc[t]);
    __syncthreads();
  }
  const int cl = lane & 15, rofs = (lane >> 4) << 3;
#pragma unroll
  for (int t = 0; t < 4; ++t) {
    int col = col0 + t * 16 + cl;
    int which = col >> 10;            // 0=q 1=k 2=v
    int hd = col & 1023;
    int head = hd >> 6, dd = hd & 63;
    bf16_t* dst = (which == 0) ? qb : ((which == 1) ? kb : vb);
#pragma unroll
    for (int v = 0; v < 8; ++v) {
      int grow = row0 + wave * 16 + rofs + v;
      int bi = grow >> 12, ni = grow & 4095;
      dst[(((size_t)bi * Hsz + head) * Nsz + ni) * Dsz + dd] = f2bf(acc[t][v]);
    }
  }
}

// ============ Kernel 2: qa_sim + softmax(dim m) -> normalized bf16 [b,h,n,m] ============
__global__ void __launch_bounds__(128)
k_qa(const bf16_t* __restrict__ qb, const float* __restrict__ agents, bf16_t* __restrict__ qa) {
  __shared__ alignas(16) bf16_t sbT[128][S72];  // agents (m x d), natural N-major for B
  __shared__ alignas(16) bf16_t sa[64][S72];    // q rows x d
  const int bh = blockIdx.x, b = bh >> 4, h = bh & 15;
  const int n0 = blockIdx.y * 64;
  const int tid = threadIdx.x, wave = tid >> 5, lane = tid & 31;
  for (int i = tid; i < Msz * Dsz; i += 128) {
    int m = i >> 6, d = i & 63;
    sbT[m][d] = f2bf(agents[((size_t)h * Msz + m) * Dsz + d] * SCALE_F);
  }
  const bf16_t* qsrc = qb + (((size_t)b * Hsz + h) * Nsz + n0) * Dsz;
  for (int i = tid; i < 64 * 8; i += 128) {     // vectorized 16B copies
    int r = i >> 3, c = i & 7;
    cp16(&sa[r][c * 8], qsrc + (size_t)r * Dsz + c * 8);
  }
  __syncthreads();
  v8f acc[8] = {};
#pragma unroll
  for (int ks = 0; ks < 2; ++ks) {
    v16bf af = frag_a(&sa[wave * 16][ks * 32], S72);
#pragma unroll
    for (int t = 0; t < 8; ++t) acc[t] = wmma_bf(af, frag_bt(&sbT[t * 16][ks * 32], S72), acc[t]);
  }
  float rinv[8];
#pragma unroll
  for (int v = 0; v < 8; ++v) {
    float mx = -3.0e38f;
#pragma unroll
    for (int t = 0; t < 8; ++t) mx = fmaxf(mx, acc[t][v]);
#pragma unroll
    for (int s = 1; s < 16; s <<= 1) mx = fmaxf(mx, __shfl_xor(mx, s, 32));
    float sm = 0.f;
#pragma unroll
    for (int t = 0; t < 8; ++t) { float e = __expf(acc[t][v] - mx); acc[t][v] = e; sm += e; }
#pragma unroll
    for (int s = 1; s < 16; s <<= 1) sm += __shfl_xor(sm, s, 32);
    rinv[v] = 1.0f / sm;
  }
  const int cl = lane & 15, rofs = (lane >> 4) << 3;
  bf16_t* dst = qa + (((size_t)b * Hsz + h) * Nsz + n0) * Msz;
#pragma unroll
  for (int t = 0; t < 8; ++t)
#pragma unroll
    for (int v = 0; v < 8; ++v)
      dst[(size_t)(wave * 16 + rofs + v) * Msz + t * 16 + cl] = f2bf(acc[t][v] * rinv[v]);
}

// ====== Kernel 3: ak_sim, mask, streaming 2-pass softmax(dim n) -> unnorm exp + 1/sum ======
__global__ void __launch_bounds__(256)
k_ak(const bf16_t* __restrict__ kbuf, const float* __restrict__ agents,
     const unsigned char* __restrict__ mask, bf16_t* __restrict__ akp,
     float* __restrict__ invsum) {
  __shared__ alignas(16) bf16_t sa[128][S72];   // agents*SCALE (m x d), A
  __shared__ alignas(16) bf16_t skT[128][S72];  // k chunk (n x d), natural N-major for B
  const int bh = blockIdx.x, b = bh >> 4, h = bh & 15;
  const int tid = threadIdx.x, wave = tid >> 5, lane = tid & 31;
  for (int i = tid; i < Msz * Dsz; i += 256) {
    int m = i >> 6, d = i & 63;
    sa[m][d] = f2bf(agents[((size_t)h * Msz + m) * Dsz + d] * SCALE_F);
  }
  __syncthreads();
  v16bf af0 = frag_a(&sa[wave * 16][0], S72);
  v16bf af1 = frag_a(&sa[wave * 16][32], S72);
  const bf16_t* ksrc = kbuf + ((size_t)b * Hsz + h) * (size_t)Nsz * Dsz;
  const unsigned char* mrow = mask + (size_t)b * Nsz;
  const int cl = lane & 15, rofs = (lane >> 4) << 3;
  float rmax[8];
#pragma unroll
  for (int v = 0; v < 8; ++v) rmax[v] = -3.0e38f;
  // pass 1: row max
  for (int c = 0; c < Nsz / 128; ++c) {
    __syncthreads();
    for (int i = tid; i < 128 * 8; i += 256) {
      int n = i >> 3, c8 = i & 7;
      cp16(&skT[n][c8 * 8], ksrc + (size_t)(c * 128 + n) * Dsz + c8 * 8);
    }
    __syncthreads();
    v8f acc[8] = {};
#pragma unroll
    for (int t = 0; t < 8; ++t) {
      acc[t] = wmma_bf(af0, frag_bt(&skT[t * 16][0], S72), acc[t]);
      acc[t] = wmma_bf(af1, frag_bt(&skT[t * 16][32], S72), acc[t]);
    }
#pragma unroll
    for (int t = 0; t < 8; ++t) {
      float pen = mrow[c * 128 + t * 16 + cl] ? 0.0f : -3.0e38f;
#pragma unroll
      for (int v = 0; v < 8; ++v) rmax[v] = fmaxf(rmax[v], acc[t][v] + pen);
    }
  }
#pragma unroll
  for (int v = 0; v < 8; ++v)
#pragma unroll
    for (int s = 1; s < 16; s <<= 1) rmax[v] = fmaxf(rmax[v], __shfl_xor(rmax[v], s, 32));
  // pass 2: exp, accumulate sum, store unnormalized probs
  float rsum[8] = {};
  bf16_t* dst = akp + (size_t)bh * (size_t)Msz * Nsz;
  for (int c = 0; c < Nsz / 128; ++c) {
    __syncthreads();
    for (int i = tid; i < 128 * 8; i += 256) {
      int n = i >> 3, c8 = i & 7;
      cp16(&skT[n][c8 * 8], ksrc + (size_t)(c * 128 + n) * Dsz + c8 * 8);
    }
    __syncthreads();
    v8f acc[8] = {};
#pragma unroll
    for (int t = 0; t < 8; ++t) {
      acc[t] = wmma_bf(af0, frag_bt(&skT[t * 16][0], S72), acc[t]);
      acc[t] = wmma_bf(af1, frag_bt(&skT[t * 16][32], S72), acc[t]);
    }
#pragma unroll
    for (int t = 0; t < 8; ++t) {
      bool ok = mrow[c * 128 + t * 16 + cl] != 0;
#pragma unroll
      for (int v = 0; v < 8; ++v) {
        float p = ok ? __expf(acc[t][v] - rmax[v]) : 0.0f;
        rsum[v] += p;
        dst[(size_t)(wave * 16 + rofs + v) * Nsz + c * 128 + t * 16 + cl] = f2bf(p);
      }
    }
  }
#pragma unroll
  for (int v = 0; v < 8; ++v) {
#pragma unroll
    for (int s = 1; s < 16; s <<= 1) rsum[v] += __shfl_xor(rsum[v], s, 32);
    if (cl == 0) invsum[(size_t)bh * Msz + wave * 16 + rofs + v] = 1.0f / rsum[v];
  }
}

// ====== Kernel 4: talking-heads mix via WMMA: D[jj,g] = P'[jj,h] x W^T[h,g], in place ======
// p layout [b,h,plane] with plane = N*M flat; each wave owns disjoint 16-position tiles.
// use_inv: fold per-(b,h,m) 1/rowsum into A-staging (ak path; plane = m*N + n, N%16==0).
__global__ void __launch_bounds__(256)
k_mix_w(bf16_t* __restrict__ p, const float* __restrict__ w,
        const float* __restrict__ invsum, int use_inv) {
  __shared__ alignas(16) bf16_t sB[16][S40];        // W row-major == N-major B, K 16..31 zero
  __shared__ alignas(16) bf16_t sA[8][16][S40];     // per-wave A tile, K 16..31 zero
  const int tid = threadIdx.x, wave = tid >> 5, lane = tid & 31;
  {                                                  // stage B = W (zero-pad K)
    int g = tid >> 4, hh = tid & 15;
    sB[g][hh + 16] = f2bf(0.0f);
    sB[g][hh] = f2bf(w[tid]);
  }
  for (int i = lane; i < 256; i += 32)               // zero this wave's A upper-K half
    sA[wave][i >> 4][16 + (i & 15)] = f2bf(0.0f);
  __syncthreads();
  const v16bf bw = frag_bt(&sB[0][0], S40);
  const size_t plane = (size_t)Nsz * Msz;            // 524288
  const size_t block_base = (size_t)blockIdx.x * 2048;
  const int b = (int)(block_base / plane);
  const size_t inplane0 = block_base % plane;
  bf16_t* pb = p + (size_t)b * Hsz * plane;
  const int cl = lane & 15, rofs = (lane >> 4) << 3;
  for (int t = 0; t < 16; ++t) {
    const size_t jj0 = inplane0 + (size_t)wave * 256 + (size_t)t * 16;
    const int m = (int)(jj0 / (size_t)Nsz);          // uniform over the 16-tile (ak path)
#pragma unroll
    for (int e = 0; e < 8; ++e) {                    // stage A: sA[jj][h] = p[h][jj0+jj]
      int i = lane + 32 * e;
      int hh = i >> 4, jj = i & 15;
      float val = bf2f(pb[(size_t)hh * plane + jj0 + jj]);
      if (use_inv) val *= invsum[((size_t)b * Hsz + hh) * Msz + m];
      sA[wave][jj][hh] = f2bf(val);
    }
    v16bf af = frag_a(&sA[wave][0][0], S40);
    v8f acc = {};
    acc = wmma_bf(af, bw, acc);
#pragma unroll
    for (int v = 0; v < 8; ++v)                      // D: col g = cl, row jj = rofs+v
      pb[(size_t)cl * plane + jj0 + rofs + v] = f2bf(acc[v]);
  }
}

// ====== Kernel 5: agent_out[b,h,m,d] = ak_mix @ v  (K = n = 4096) ======
__global__ void __launch_bounds__(256)
k_agent_out(const bf16_t* __restrict__ ak, const bf16_t* __restrict__ vb, bf16_t* __restrict__ aout) {
  __shared__ alignas(16) bf16_t sa[128][S40];   // ak rows m x K(32)
  __shared__ alignas(16) bf16_t svT[64][S40];   // v chunk transposed: [d][k]
  const int bh = blockIdx.x;
  const int tid = threadIdx.x, wave = tid >> 5, lane = tid & 31;
  const bf16_t* asrc = ak + (size_t)bh * (size_t)Msz * Nsz;
  const bf16_t* vsrc = vb + (size_t)bh * (size_t)Nsz * Dsz;
  v8f acc[4] = {};
  for (int k0 = 0; k0 < Nsz; k0 += 32) {
    __syncthreads();
    for (int i = tid; i < 128 * 4; i += 256) {
      int m = i >> 2, c8 = i & 3;
      cp16(&sa[m][c8 * 8], asrc + (size_t)m * Nsz + k0 + c8 * 8);
    }
    for (int i = tid; i < 32 * 64; i += 256) {   // coalesced over d, transpose into LDS
      int kk = i >> 6, d = i & 63;
      svT[d][kk] = vsrc[(size_t)(k0 + kk) * Dsz + d];
    }
    __syncthreads();
    v16bf af = frag_a(&sa[wave * 16][0], S40);
#pragma unroll
    for (int t = 0; t < 4; ++t) acc[t] = wmma_bf(af, frag_bt(&svT[t * 16][0], S40), acc[t]);
  }
  const int cl = lane & 15, rofs = (lane >> 4) << 3;
  bf16_t* dst = aout + (size_t)bh * Msz * Dsz;
#pragma unroll
  for (int t = 0; t < 4; ++t)
#pragma unroll
    for (int v = 0; v < 8; ++v)
      dst[(size_t)(wave * 16 + rofs + v) * Dsz + t * 16 + cl] = f2bf(acc[t][v]);
}

// ====== Kernel 6: out = qa_mix @ agent_out, query-mask, store [b,n,h*64+d] bf16 ======
__global__ void __launch_bounds__(128)
k_out2(const bf16_t* __restrict__ qa, const bf16_t* __restrict__ aout,
       const unsigned char* __restrict__ mask, bf16_t* __restrict__ attno) {
  __shared__ alignas(16) bf16_t sbT[64][S136];  // agent_out transposed: [d][m]
  __shared__ alignas(16) bf16_t sa[64][S136];   // qa rows n x m
  const int bh = blockIdx.x, b = bh >> 4, h = bh & 15;
  const int n0 = blockIdx.y * 64;
  const int tid = threadIdx.x, wave = tid >> 5, lane = tid & 31;
  const bf16_t* bs = aout + (size_t)bh * Msz * Dsz;
  for (int i = tid; i < Msz * Dsz; i += 128) {  // coalesced over d, transpose
    int m = i >> 6, d = i & 63;
    sbT[d][m] = bs[i];
  }
  const bf16_t* as = qa + ((size_t)bh * Nsz + n0) * Msz;
  for (int i = tid; i < 64 * 16; i += 128) {    // vectorized copies
    int r = i >> 4, c = i & 15;
    cp16(&sa[r][c * 8], as + (size_t)r * Msz + c * 8);
  }
  __syncthreads();
  v8f acc[4] = {};
#pragma unroll
  for (int ks = 0; ks < 4; ++ks) {
    v16bf af = frag_a(&sa[wave * 16][ks * 32], S136);
#pragma unroll
    for (int t = 0; t < 4; ++t) acc[t] = wmma_bf(af, frag_bt(&sbT[t * 16][ks * 32], S136), acc[t]);
  }
  const int cl = lane & 15, rofs = (lane >> 4) << 3;
#pragma unroll
  for (int v = 0; v < 8; ++v) {
    int n = n0 + wave * 16 + rofs + v;
    float mval = mask[(size_t)b * Nsz + n] ? 1.0f : 0.0f;
#pragma unroll
    for (int t = 0; t < 4; ++t)
      attno[((size_t)b * Nsz + n) * DINNER + h * Dsz + t * 16 + cl] = f2bf(acc[t][v] * mval);
  }
}

// ====== Kernel 7: final projection attn_out @ W_out -> f32 d_out ======
__global__ void __launch_bounds__(128)
k_proj(const bf16_t* __restrict__ attno, const float* __restrict__ wout, float* __restrict__ out) {
  __shared__ alignas(16) bf16_t sx[64][S40];
  __shared__ alignas(16) bf16_t swT[64][S40];
  const int row0 = blockIdx.x * 64, col0 = blockIdx.y * 64;
  const int tid = threadIdx.x, wave = tid >> 5, lane = tid & 31;
  v8f acc[4] = {};
  for (int k0 = 0; k0 < DINNER; k0 += 32) {
    __syncthreads();
    for (int i = tid; i < 64 * 4; i += 128) {
      int r = i >> 2, c8 = i & 3;
      cp16(&sx[r][c8 * 8], attno + (size_t)(row0 + r) * DINNER + k0 + c8 * 8);
    }
    for (int i = tid; i < 32 * 64; i += 128) {
      int r = i >> 6, c = i & 63;
      swT[c][r] = f2bf(wout[(size_t)(k0 + r) * DIMsz + col0 + c]);
    }
    __syncthreads();
    v16bf af = frag_a(&sx[wave * 16][0], S40);
#pragma unroll
    for (int t = 0; t < 4; ++t) acc[t] = wmma_bf(af, frag_bt(&swT[t * 16][0], S40), acc[t]);
  }
  const int cl = lane & 15, rofs = (lane >> 4) << 3;
#pragma unroll
  for (int t = 0; t < 4; ++t)
#pragma unroll
    for (int v = 0; v < 8; ++v)
      out[(size_t)(row0 + wave * 16 + rofs + v) * DIMsz + col0 + t * 16 + cl] = acc[t][v];
}

extern "C" void kernel_launch(void* const* d_in, const int* in_sizes, int n_in,
                              void* d_out, int out_size, void* d_ws, size_t ws_size,
                              hipStream_t stream) {
  (void)in_sizes; (void)n_in; (void)out_size; (void)ws_size;
  const float* x              = (const float*)d_in[0];
  const unsigned char* mask   = (const unsigned char*)d_in[1];
  const float* wqkv           = (const float*)d_in[2];
  const float* agents         = (const float*)d_in[3];
  const float* wqa            = (const float*)d_in[4];
  const float* wak            = (const float*)d_in[5];
  const float* wout           = (const float*)d_in[6];
  float* out = (float*)d_out;

  char* ws = (char*)d_ws;
  size_t off = 0;
  auto alloc = [&](size_t bytes) -> void* {
    void* p = ws + off;
    off += (bytes + 255) & ~(size_t)255;
    return p;
  };
  const size_t QKVELEM = (size_t)Bsz * Hsz * Nsz * Dsz;
  bf16_t* qb    = (bf16_t*)alloc(QKVELEM * 2);
  bf16_t* kb    = (bf16_t*)alloc(QKVELEM * 2);
  bf16_t* vb    = (bf16_t*)alloc(QKVELEM * 2);
  bf16_t* qa    = (bf16_t*)alloc((size_t)Bsz * Hsz * Nsz * Msz * 2);
  bf16_t* akp   = (bf16_t*)alloc((size_t)Bsz * Hsz * Msz * Nsz * 2);
  float*  invs  = (float*)alloc((size_t)Bsz * Hsz * Msz * 4);
  bf16_t* aout  = (bf16_t*)alloc((size_t)Bsz * Hsz * Msz * Dsz * 2);
  bf16_t* attno = (bf16_t*)alloc((size_t)Bsz * Nsz * DINNER * 2);

  dim3 g1((Bsz * Nsz) / 64, QKV3 / 64);           // 256 x 48
  k_qkv<<<g1, 128, 0, stream>>>(x, wqkv, qb, kb, vb);

  dim3 g2(Bsz * Hsz, Nsz / 64);                   // 64 x 64
  k_qa<<<g2, 128, 0, stream>>>(qb, agents, qa);

  k_ak<<<Bsz * Hsz, 256, 0, stream>>>(kb, agents, mask, akp, invs);

  // talking-heads mixing via WMMA, in place (2048 positions per block)
  int mixblocks = (int)(((size_t)Bsz * Nsz * Msz) / 2048);   // 1024
  k_mix_w<<<mixblocks, 256, 0, stream>>>(qa, wqa, nullptr, 0);
  k_mix_w<<<mixblocks, 256, 0, stream>>>(akp, wak, invs, 1);

  k_agent_out<<<Bsz * Hsz, 256, 0, stream>>>(akp, vb, aout);

  dim3 g7(Bsz * Hsz, Nsz / 64);
  k_out2<<<g7, 128, 0, stream>>>(qa, aout, mask, attno);

  dim3 g8((Bsz * Nsz) / 64, DIMsz / 64);          // 256 x 16
  k_proj<<<g8, 128, 0, stream>>>(attno, wout, out);
}
